// LSTM_Model_63582695850543
// MI455X (gfx1250) — matrix operand order repaired and checked by reading, not compiled, not run on hardware
//
#include <hip/hip_runtime.h>
#include <math.h>

typedef __attribute__((ext_vector_type(16))) _Float16 v16h;
typedef __attribute__((ext_vector_type(8)))  _Float16 v8h;
typedef __attribute__((ext_vector_type(8)))  float    v8f;

#define LSTM_B  16384
#define LSTM_T  200
#define LSTM_IN 9
#define LSTM_H  64
#define LSTM_G  256   // 4*H
#define HSTR    72    // padded row stride (halfs) for h buffers: kills LDS bank conflicts

// LDS layout (units = halfs).  Weight fragments are pre-swizzled into the exact
// per-lane B-operand layout of v_wmma_f32_16x16x32_f16:
//   frag(tn, kc): lane L holds column N = 16*tn + (L&15),
//                 half i = K = 32*kc + (i>=8 ? 16 : 0) + (L>=16 ? 8 : 0) + (i&7)
#define OFF_WIH1 0                       // 16 tiles * 1 kchunk * 512 = 8192
#define OFF_WHH1 8192                    // 16 * 2 * 512 = 16384
#define OFF_WIH2 24576                   // 16384
#define OFF_WHH2 40960                   // 16384
#define OFF_HBUF 57344                   // 4 waves * 2 layers * 16 rows * HSTR
#define TOT_HALFS (OFF_HBUF + 4 * 2 * 16 * HSTR)   // 66560 halfs = 133120 B

#if __has_builtin(__builtin_amdgcn_tanhf)
#define FAST_TANH(x) __builtin_amdgcn_tanhf(x)   // v_tanh_f32 (gfx1250 TRANS op)
#else
#define FAST_TANH(x) tanhf(x)
#endif

__device__ __forceinline__ float sigf(float x) {
  // sigmoid(x) == 0.5*tanh(0.5x) + 0.5, single transcendental
  return fmaf(0.5f, FAST_TANH(0.5f * x), 0.5f);
}

#define WMMA16(a, b, c) \
  __builtin_amdgcn_wmma_f32_16x16x32_f16(false, (a), false, (b), (short)0, (c), false, false)

// B-operand fragment: 16 contiguous halfs per lane (two ds_load_b128)
__device__ __forceinline__ v16h load_bfrag(const _Float16* base, int nkc, int tn, int kc, int lane) {
  const _Float16* p = base + (((tn * nkc + kc) * 32) + lane) * 16;
  v8h lo = *(const v8h*)(p);
  v8h hi = *(const v8h*)(p + 8);
  return __builtin_shufflevector(lo, hi, 0,1,2,3,4,5,6,7,8,9,10,11,12,13,14,15);
}

// A-operand fragment for a 16x64 f16 activation tile stored row-major (stride HSTR)
// ISA A layout: half i -> K = 32*kc + (i<8 ? i : (i-8)+16) + 8*(lane>=16)
// => two contiguous 8-half (16B) runs per lane.
__device__ __forceinline__ v16h load_afrag(const _Float16* hb, int m, int dlt, int kc) {
  const _Float16* p = hb + m * HSTR + 32 * kc + 8 * dlt;
  v8h lo = *(const v8h*)(p);
  v8h hi = *(const v8h*)(p + 16);
  return __builtin_shufflevector(lo, hi, 0,1,2,3,4,5,6,7,8,9,10,11,12,13,14,15);
}

// Convert W (row-major [G x Kdim] f32) into swizzled f16 B-fragments in LDS.
__device__ __forceinline__ void stage_weight(_Float16* dst, const float* W, int Kdim, int nkc, int tid) {
  int total = 16 * nkc * 512;
  for (int idx = tid; idx < total; idx += 128) {
    int i    = idx & 15;
    int lane = (idx >> 4) & 31;
    int kc   = (idx >> 9) % nkc;
    int tn   = idx / (512 * nkc);
    int k = 32 * kc + ((i & 8) ? 16 : 0) + (i & 7) + ((lane & 16) ? 8 : 0);
    int g = 16 * tn + (lane & 15);
    dst[idx] = (k < Kdim) ? (_Float16)W[g * Kdim + k] : (_Float16)0.0f;
  }
}

__global__ __launch_bounds__(128)
void lstm2_wmma_kernel(const float* __restrict__ x,
                       const float* __restrict__ Wih1, const float* __restrict__ Whh1,
                       const float* __restrict__ bih1, const float* __restrict__ bhh1,
                       const float* __restrict__ Wih2, const float* __restrict__ Whh2,
                       const float* __restrict__ bih2, const float* __restrict__ bhh2,
                       const float* __restrict__ Wout, const float* __restrict__ bout,
                       float* __restrict__ out)
{
  extern __shared__ _Float16 smem[];
  const int tid  = threadIdx.x;
  const int lane = tid & 31;
  const int wave = tid >> 5;
  const int ln   = lane & 15;          // row (A) or column (B/C/D) within tile
  const int dlt  = (lane >> 4) & 1;    // upper half-wave selector

  // ---- one-time staging: weights -> swizzled f16 fragments in LDS ----
  stage_weight(smem + OFF_WIH1, Wih1, LSTM_IN, 1, tid);
  stage_weight(smem + OFF_WHH1, Whh1, LSTM_H,  2, tid);
  stage_weight(smem + OFF_WIH2, Wih2, LSTM_H,  2, tid);
  stage_weight(smem + OFF_WHH2, Whh2, LSTM_H,  2, tid);

  _Float16* h1b = smem + OFF_HBUF + wave * (2 * 16 * HSTR);
  _Float16* h2b = h1b + 16 * HSTR;
  for (int i2 = lane; i2 < 2 * 16 * HSTR; i2 += 32) h1b[i2] = (_Float16)0.0f;
  __syncthreads();

  // per-lane fused biases: D-layout lane holds gate column g = 16*tn + ln
  float bias1v[16], bias2v[16];
  #pragma unroll
  for (int tn = 0; tn < 16; ++tn) {
    int g = 16 * tn + ln;
    bias1v[tn] = bih1[g] + bhh1[g];
    bias2v[tn] = bih2[g] + bhh2[g];
  }

  // cell states in registers, C/D fragment layout: c[hn][v] is hidden unit
  // nh = 16*hn + ln, batch row m = v + 8*dlt
  float c1[4][8], c2[4][8];
  #pragma unroll
  for (int a = 0; a < 4; ++a)
    #pragma unroll
    for (int v = 0; v < 8; ++v) { c1[a][v] = 0.0f; c2[a][v] = 0.0f; }

  const int rowBase = blockIdx.x * 64 + wave * 16;
  const float* xrow = x + (size_t)(rowBase + ln) * (LSTM_T * LSTM_IN);

  for (int t = 0; t < LSTM_T; ++t) {
    // ---- A fragment from x_t (K = 9, zero padded to 32) ----
    const float* xp = xrow + t * LSTM_IN;
    v16h ax = {};
    if (dlt == 0) {
      #pragma unroll
      for (int j = 0; j < 8; ++j) ax[j] = (_Float16)xp[j];   // K = 0..7
    } else {
      ax[0] = (_Float16)xp[8];                               // K = 8
    }
    if (t + 1 < LSTM_T) __builtin_prefetch(xp + LSTM_IN, 0, 1);  // global_prefetch_b8

    // ensure previous step's h stores are visible, load old h1 as A operand
    asm volatile("s_wait_dscnt 0x0" ::: "memory");
    v16h a1_0 = load_afrag(h1b, ln, dlt, 0);
    v16h a1_1 = load_afrag(h1b, ln, dlt, 1);

    // ---- layer 1: gates = x@Wih1^T + h1@Whh1^T + b ----
    #pragma unroll
    for (int hn = 0; hn < 4; ++hn) {
      v8f acc[4];
      #pragma unroll
      for (int gi = 0; gi < 4; ++gi) {          // i, f, g, o
        int tn = 4 * gi + hn;
        float bv = bias1v[tn];
        v8f c = {bv, bv, bv, bv, bv, bv, bv, bv};
        c = WMMA16(ax,   load_bfrag(smem + OFF_WIH1, 1, tn, 0, lane), c);
        c = WMMA16(a1_0, load_bfrag(smem + OFF_WHH1, 2, tn, 0, lane), c);
        c = WMMA16(a1_1, load_bfrag(smem + OFF_WHH1, 2, tn, 1, lane), c);
        acc[gi] = c;
      }
      #pragma unroll
      for (int v = 0; v < 8; ++v) {
        float ig = sigf(acc[0][v]);
        float fg = sigf(acc[1][v]);
        float gg = FAST_TANH(acc[2][v]);
        float og = sigf(acc[3][v]);
        float cc = fg * c1[hn][v] + ig * gg;
        c1[hn][v] = cc;
        float hh = og * FAST_TANH(cc);
        h1b[(v + 8 * dlt) * HSTR + 16 * hn + ln] = (_Float16)hh;   // transpose via LDS
      }
    }

    // ---- layer 2: gates = h1_new@Wih2^T + h2_old@Whh2^T + b ----
    asm volatile("s_wait_dscnt 0x0" ::: "memory");
    v16h b1_0 = load_afrag(h1b, ln, dlt, 0);
    v16h b1_1 = load_afrag(h1b, ln, dlt, 1);
    v16h a2_0 = load_afrag(h2b, ln, dlt, 0);
    v16h a2_1 = load_afrag(h2b, ln, dlt, 1);

    #pragma unroll
    for (int hn = 0; hn < 4; ++hn) {
      v8f acc[4];
      #pragma unroll
      for (int gi = 0; gi < 4; ++gi) {
        int tn = 4 * gi + hn;
        float bv = bias2v[tn];
        v8f c = {bv, bv, bv, bv, bv, bv, bv, bv};
        c = WMMA16(b1_0, load_bfrag(smem + OFF_WIH2, 2, tn, 0, lane), c);
        c = WMMA16(b1_1, load_bfrag(smem + OFF_WIH2, 2, tn, 1, lane), c);
        c = WMMA16(a2_0, load_bfrag(smem + OFF_WHH2, 2, tn, 0, lane), c);
        c = WMMA16(a2_1, load_bfrag(smem + OFF_WHH2, 2, tn, 1, lane), c);
        acc[gi] = c;
      }
      #pragma unroll
      for (int v = 0; v < 8; ++v) {
        float ig = sigf(acc[0][v]);
        float fg = sigf(acc[1][v]);
        float gg = FAST_TANH(acc[2][v]);
        float og = sigf(acc[3][v]);
        float cc = fg * c2[hn][v] + ig * gg;
        c2[hn][v] = cc;
        float hh = og * FAST_TANH(cc);
        h2b[(v + 8 * dlt) * HSTR + 16 * hn + ln] = (_Float16)hh;
      }
    }
  }

  // ---- epilogue: out = h2 @ Wout^T + bout  (tiny: 3 outputs) ----
  asm volatile("s_wait_dscnt 0x0" ::: "memory");
  if (lane < 16) {
    #pragma unroll
    for (int o = 0; o < 3; ++o) {
      float s = bout[o];
      for (int k = 0; k < LSTM_H; ++k)
        s += (float)h2b[lane * HSTR + k] * Wout[o * LSTM_H + k];
      out[(size_t)(rowBase + lane) * 3 + o] = s;
    }
  }
}

extern "C" void kernel_launch(void* const* d_in, const int* in_sizes, int n_in,
                              void* d_out, int out_size, void* d_ws, size_t ws_size,
                              hipStream_t stream) {
  (void)in_sizes; (void)n_in; (void)out_size; (void)d_ws; (void)ws_size;
  const float* x    = (const float*)d_in[0];
  const float* Wih1 = (const float*)d_in[1];
  const float* Whh1 = (const float*)d_in[2];
  const float* bih1 = (const float*)d_in[3];
  const float* bhh1 = (const float*)d_in[4];
  const float* Wih2 = (const float*)d_in[5];
  const float* Whh2 = (const float*)d_in[6];
  const float* bih2 = (const float*)d_in[7];
  const float* bhh2 = (const float*)d_in[8];
  const float* Wout = (const float*)d_in[9];
  const float* bout = (const float*)d_in[10];
  float* out = (float*)d_out;

  constexpr size_t smem_bytes = (size_t)TOT_HALFS * sizeof(_Float16);  // 133120 B < 320 KB
  (void)hipFuncSetAttribute((const void*)lstm2_wmma_kernel,
                            hipFuncAttributeMaxDynamicSharedMemorySize, (int)smem_bytes);
  dim3 grid(LSTM_B / 64), block(128);
  lstm2_wmma_kernel<<<grid, block, smem_bytes, stream>>>(
      x, Wih1, Whh1, bih1, bhh1, Wih2, Whh2, bih2, bhh2, Wout, bout, out);
}
